// SSMBlock_26250840113810
// MI455X (gfx1250) — matrix operand order repaired and checked by reading, not compiled, not run on hardware
//
#include <hip/hip_runtime.h>
#include <hip/hip_bf16.h>

typedef __attribute__((ext_vector_type(16))) __bf16 v16bf;
typedef __attribute__((ext_vector_type(8)))  __bf16 v8bf;
typedef __attribute__((ext_vector_type(8)))  float  v8f;
typedef __attribute__((ext_vector_type(4)))  unsigned int v4u;
typedef __attribute__((ext_vector_type(8)))  int v8i;
typedef __attribute__((ext_vector_type(4)))  int v4i;

// Problem constants
#define Bc 2
#define Lc 1024
#define Hc 1024
#define Dc 2048
#define Nc 16
#define Kc 4
#define Rc 64
#define Pc 96           // R + 2N
#define Mc (Bc*Lc)      // 2048 rows

// GEMM tiling
#define BM 128
#define BN 64
#define BK 32

// ---------------------------------------------------------------------------
// TDM: async 2-D tile DMA global -> LDS (D# packed per CDNA5 ISA 08, §8.3/8.4)
// 2-byte elements; tile_d0 x tile_d1 tile, row stride = stride0 elements.
// Writes LDS contiguously (X fastest) == As[row][col] with row stride tile_d0.
// ---------------------------------------------------------------------------
__device__ __forceinline__ void tdm_load_tile_2d_b16(
    unsigned lds_off, const void* gptr,
    unsigned tensor_d0, unsigned tensor_d1,
    unsigned tile_d0, unsigned tile_d1, unsigned stride0)
{
    unsigned long long ga = (unsigned long long)(uintptr_t)gptr;
    v4u g0;
    g0[0] = 1u;                                               // count=1 (valid), user mode
    g0[1] = lds_off;                                          // lds_addr (bytes)
    g0[2] = (unsigned)(ga & 0xFFFFFFFFu);                     // global_addr[31:0]
    g0[3] = (unsigned)((ga >> 32) & 0x01FFFFFFu) | (2u << 30); // addr[56:32] | type=2
    v8i g1;
    g1[0] = (int)(1u << 16);                                  // data_size=1 (2B); no multicast/pad
    g1[1] = (int)((tensor_d0 & 0xFFFFu) << 16);               // tensor_dim0[15:0] @ [63:48]
    g1[2] = (int)(((tensor_d0 >> 16) & 0xFFFFu) |             // tensor_dim0[31:16]
                  ((tensor_d1 & 0xFFFFu) << 16));             // tensor_dim1[15:0]
    g1[3] = (int)(((tensor_d1 >> 16) & 0xFFFFu) |             // tensor_dim1[31:16]
                  ((tile_d0 & 0xFFFFu) << 16));               // tile_dim0
    g1[4] = (int)(tile_d1 & 0xFFFFu);                         // tile_dim1; tile_dim2=0
    g1[5] = (int)stride0;                                     // tensor_dim0_stride[31:0]
    g1[6] = 0;                                                // stride0[47:32]=0; stride1[15:0]=0
    g1[7] = 0;
    v4i gz = {0, 0, 0, 0};
#if defined(__clang_major__) && (__clang_major__ >= 23)
    v8i gz8 = {0, 0, 0, 0, 0, 0, 0, 0};
    __builtin_amdgcn_tensor_load_to_lds(g0, g1, gz, gz, gz8, 0);
#else
    __builtin_amdgcn_tensor_load_to_lds(g0, g1, gz, gz, 0);
#endif
}

// ---------------------------------------------------------------------------
// Generic bf16 WMMA GEMM: C[M,N] = A[M,K](bf16) * W[N,K](bf16)^T + bias [+res]
// A tile staged by the Tensor Data Mover (async, TENSORcnt); W tile staged by
// vector loads + transposed LDS stores (WMMA B layout needs K-in-lane).
// ---------------------------------------------------------------------------
__global__ __launch_bounds__(256)
void gemm_wmma_bf16_kernel(const __bf16* __restrict__ A, int lda,
                           const __bf16* __restrict__ W, int ldw,
                           const float* __restrict__ bias,
                           const float* __restrict__ residual,   // ld = ldc, nullable
                           float* __restrict__ Cf,               // nullable
                           __bf16* __restrict__ Cb,              // nullable
                           int ldc, int M, int N, int K)
{
    __shared__ __attribute__((aligned(32))) __bf16 As[BM * BK];   // [m][k]
    __shared__ __attribute__((aligned(32))) __bf16 Bs[BK * BN];   // [k][n] (transposed W)

    const int tid  = threadIdx.x;
    const int lane = tid & 31;
    const int wv   = tid >> 5;       // 0..7
    const int wm   = wv >> 1;        // 0..3  -> 32-row strip
    const int wn   = wv & 1;         // 0..1  -> 32-col strip
    const int m_blk = blockIdx.x * BM;
    const int n_blk = blockIdx.y * BN;

    const unsigned As_off = (unsigned)(uintptr_t)(void*)As;

    v8f acc[2][2] = {};

    // W global->LDS mapping
    const int b_n   = tid >> 2;            // 0..63
    const int b_kg  = (tid & 3) * 8;       // 0,8,16,24

    const int khalf = lane >> 4;           // A-fragment K half select
    const int ml    = lane & 15;

    for (int k0 = 0; k0 < K; k0 += BK) {
        // ---- stage A tile (128x32 bf16) via TDM; wave 0 issues one descriptor
        if (tid < 32) {
            tdm_load_tile_2d_b16(As_off,
                                 A + (size_t)m_blk * lda + k0,
                                 (unsigned)K, (unsigned)M,
                                 (unsigned)BK, (unsigned)BM, (unsigned)lda);
            __builtin_amdgcn_s_wait_tensorcnt(0);
        }
        // ---- stage W tile transposed into Bs[k][n]
        {
            __bf16 bv[8];
            if (n_blk + b_n < N) {
                const __bf16* wg = W + (size_t)(n_blk + b_n) * ldw + (k0 + b_kg);
                *(uint4*)bv = *(const uint4*)wg;
                if (k0 + BK < K) __builtin_prefetch(wg + BK, 0, 1);
            } else {
                #pragma unroll
                for (int i = 0; i < 8; ++i) bv[i] = (__bf16)0.0f;
            }
            #pragma unroll
            for (int i = 0; i < 8; ++i) Bs[(b_kg + i) * BN + b_n] = bv[i];
        }
        __syncthreads();

        #pragma unroll
        for (int mi = 0; mi < 2; ++mi) {
            const int mrow = wm * 32 + mi * 16 + ml;
            // A fragment per ISA layout: lanes 0-15 hold K 0-7 & 16-23, lanes 16-31 hold K 8-15 & 24-31
            v8bf lo = *(const v8bf*)&As[mrow * BK + khalf * 8];
            v8bf hi = *(const v8bf*)&As[mrow * BK + 16 + khalf * 8];
            v16bf afrag;
            #pragma unroll
            for (int i = 0; i < 8; ++i) { afrag[i] = lo[i]; afrag[i + 8] = hi[i]; }
            #pragma unroll
            for (int ni = 0; ni < 2; ++ni) {
                const int ncol = wn * 32 + ni * 16;
                // B fragment: lane&15 = N column, K in same order as A fragment
                v16bf bfrag;
                #pragma unroll
                for (int i = 0; i < 8; ++i) {
                    bfrag[i]     = Bs[(khalf * 8 + i) * BN + ncol + ml];
                    bfrag[i + 8] = Bs[(16 + khalf * 8 + i) * BN + ncol + ml];
                }
                acc[mi][ni] = __builtin_amdgcn_wmma_f32_16x16x32_bf16(
                    false, afrag, false, bfrag, (short)0, acc[mi][ni], false, false);
            }
        }
        __syncthreads();
    }

    // ---- epilogue: C layout lanes 0-15 -> M=r, lanes 16-31 -> M=8+r; N = lane&15
    #pragma unroll
    for (int mi = 0; mi < 2; ++mi) {
        #pragma unroll
        for (int ni = 0; ni < 2; ++ni) {
            const int nc = n_blk + wn * 32 + ni * 16 + (lane & 15);
            if (nc < N) {
                const float bsv = bias ? bias[nc] : 0.0f;
                #pragma unroll
                for (int r = 0; r < 8; ++r) {
                    const int mr = m_blk + wm * 32 + mi * 16 + (lane >> 4) * 8 + r;
                    float v = acc[mi][ni][r] + bsv;
                    if (residual) v += residual[(size_t)mr * ldc + nc];
                    if (Cf) Cf[(size_t)mr * ldc + nc] = v;
                    if (Cb) Cb[(size_t)mr * ldc + nc] = (__bf16)v;
                }
            }
        }
    }
}

// ---------------------------------------------------------------------------
// fp32 -> bf16 conversion
// ---------------------------------------------------------------------------
__global__ __launch_bounds__(256)
void f32_to_bf16_kernel(const float* __restrict__ src, __bf16* __restrict__ dst, int n)
{
    int i = blockIdx.x * 256 + threadIdx.x;
    if (i < n) dst[i] = (__bf16)src[i];
}

// ---------------------------------------------------------------------------
// RMSNorm over H, bf16 output (one block per row)
// ---------------------------------------------------------------------------
__global__ __launch_bounds__(256)
void rmsnorm_kernel(const float* __restrict__ x, const float* __restrict__ w,
                    __bf16* __restrict__ out)
{
    const int row = blockIdx.x;
    const float* xr = x + (size_t)row * Hc;
    float s = 0.0f;
    for (int i = threadIdx.x; i < Hc; i += 256) { float v = xr[i]; s += v * v; }
    __shared__ float red[256];
    red[threadIdx.x] = s;
    __syncthreads();
    for (int off = 128; off > 0; off >>= 1) {
        if (threadIdx.x < off) red[threadIdx.x] += red[threadIdx.x + off];
        __syncthreads();
    }
    const float rs = rsqrtf(red[0] / (float)Hc + 1e-6f);
    for (int i = threadIdx.x; i < Hc; i += 256)
        out[(size_t)row * Hc + i] = (__bf16)(xr[i] * rs * w[i]);
}

// ---------------------------------------------------------------------------
// Depthwise causal conv (K=4) + SiLU; reads x half of xg[M,2D], writes f32+bf16
// ---------------------------------------------------------------------------
__global__ __launch_bounds__(256)
void conv_silu_kernel(const float* __restrict__ xg, const float* __restrict__ cw,
                      const float* __restrict__ cb, float* __restrict__ xf,
                      __bf16* __restrict__ xb)
{
    int idx = blockIdx.x * 256 + threadIdx.x;      // (b*L + l)*D + d
    if (idx >= Mc * Dc) return;
    const int d  = idx & (Dc - 1);
    const int bl = idx >> 11;
    const int l  = bl & (Lc - 1);
    const int b  = bl >> 10;

    float acc = cb[d];
    #pragma unroll
    for (int k = 0; k < Kc; ++k) {
        const int li = l - (Kc - 1) + k;
        if (li >= 0)
            acc += xg[(size_t)(b * Lc + li) * (2 * Dc) + d] * cw[d * Kc + k];
    }
    const float s = acc / (1.0f + __expf(-acc));   // SiLU
    xf[(size_t)bl * Dc + d] = s;
    xb[(size_t)bl * Dc + d] = (__bf16)s;
}

// ---------------------------------------------------------------------------
// Selective scan: one thread per (b,d) channel; emulates the reference's
// shifted/clipped cumsum formulation exactly:
//   u[l]  = da_{l-1} * u[l-1] + dBx[l] * min(1, P[l-1]/1e-10)
//   P[l]  = P[l-1] * da_l ; hs[l] = s0 * P[l]
//   h[l]  = u[l] + hs[l]
// Fuses D-skip, SiLU(gate) and bf16 output.
// ---------------------------------------------------------------------------
__global__ __launch_bounds__(256)
void scan_kernel(const float* __restrict__ dlin,   // (M,D) pre-softplus
                 const float* __restrict__ xc,     // (M,D)
                 const float* __restrict__ xg,     // (M,2D); gate at +D
                 const float* __restrict__ proj,   // (M,96); B at 64, C at 80
                 const float* __restrict__ A_log,  // (D,16)
                 const float* __restrict__ Dp,     // (D)
                 const float* __restrict__ s0,     // (B,D,16)
                 __bf16* __restrict__ ybf)         // (M,D)
{
    const int t = blockIdx.x * 256 + threadIdx.x;
    if (t >= Bc * Dc) return;
    const int d = t & (Dc - 1);
    const int b = t >> 11;

    float Ac[Nc], u[Nc], hs[Nc], Pn[Nc], dap[Nc];
    #pragma unroll
    for (int n = 0; n < Nc; ++n) {
        Ac[n]  = -__expf(A_log[d * Nc + n]);
        u[n]   = 0.0f;
        hs[n]  = s0[((size_t)b * Dc + d) * Nc + n];
        Pn[n]  = 1.0f;
        dap[n] = 1.0f;
    }
    const float dpv = Dp[d];

    for (int l = 0; l < Lc; ++l) {
        const size_t rm = (size_t)(b * Lc + l);
        const float dl = dlin[rm * Dc + d];
        const float delta = (dl > 20.0f) ? dl : log1pf(__expf(dl));
        const float xv = xc[rm * Dc + d];
        const float g  = xg[rm * (2 * Dc) + Dc + d];
        const float dx = delta * xv;
        const float* pr = proj + rm * Pc;     // uniform across the wave
        float accv = 0.0f;
        #pragma unroll
        for (int n = 0; n < Nc; ++n) {
            const float coef = fminf(1.0f, Pn[n] * 1e10f);
            u[n] = u[n] * dap[n] + dx * pr[Rc + n] * coef;
            const float da = __expf(delta * Ac[n]);
            Pn[n] *= da;
            hs[n] *= da;
            accv += (u[n] + hs[n]) * pr[Rc + Nc + n];
            dap[n] = da;
        }
        float y = accv + xv * dpv;
        y *= g / (1.0f + __expf(-g));          // * SiLU(gate)
        ybf[rm * Dc + d] = (__bf16)y;
    }
}

// ---------------------------------------------------------------------------
// Host-side orchestration
// ---------------------------------------------------------------------------
static inline size_t align256(size_t x) { return (x + 255) & ~(size_t)255; }

extern "C" void kernel_launch(void* const* d_in, const int* in_sizes, int n_in,
                              void* d_out, int out_size, void* d_ws, size_t ws_size,
                              hipStream_t stream)
{
    (void)in_sizes; (void)n_in; (void)out_size; (void)ws_size;
    const float* inputs  = (const float*)d_in[0];
    const float* state0  = (const float*)d_in[1];
    const float* norm_w  = (const float*)d_in[2];
    const float* in_w    = (const float*)d_in[3];
    const float* in_b    = (const float*)d_in[4];
    const float* conv_w  = (const float*)d_in[5];
    const float* conv_b  = (const float*)d_in[6];
    const float* xp_w    = (const float*)d_in[7];
    const float* xp_b    = (const float*)d_in[8];
    const float* dt_w    = (const float*)d_in[9];
    const float* dt_b    = (const float*)d_in[10];
    const float* A_log   = (const float*)d_in[11];
    const float* D_param = (const float*)d_in[12];
    const float* out_w   = (const float*)d_in[13];
    const float* out_b   = (const float*)d_in[14];
    float* out = (float*)d_out;

    // workspace layout (~103 MB)
    char* ws = (char*)d_ws;
    size_t off = 0;
    __bf16* xn_bf   = (__bf16*)(ws + off); off = align256(off + (size_t)Mc * Hc * 2);
    __bf16* win_bf  = (__bf16*)(ws + off); off = align256(off + (size_t)2 * Dc * Hc * 2);
    float*  xgbuf   = (float*) (ws + off); off = align256(off + (size_t)Mc * 2 * Dc * 4);
    __bf16* xc_bf   = (__bf16*)(ws + off); off = align256(off + (size_t)Mc * Dc * 2);
    float*  xc_f    = (float*) (ws + off); off = align256(off + (size_t)Mc * Dc * 4);
    __bf16* wxp_bf  = (__bf16*)(ws + off); off = align256(off + (size_t)Pc * Dc * 2);
    float*  proj_f  = (float*) (ws + off); off = align256(off + (size_t)Mc * Pc * 4);
    __bf16* proj_bf = (__bf16*)(ws + off); off = align256(off + (size_t)Mc * Pc * 2);
    __bf16* wdt_bf  = (__bf16*)(ws + off); off = align256(off + (size_t)Dc * Rc * 2);
    float*  dlin    = (float*) (ws + off); off = align256(off + (size_t)Mc * Dc * 4);
    __bf16* y_bf    = (__bf16*)(ws + off); off = align256(off + (size_t)Mc * Dc * 2);
    __bf16* wout_bf = (__bf16*)(ws + off); off = align256(off + (size_t)Hc * Dc * 2);

    // 1) weights -> bf16
    {
        int n1 = 2 * Dc * Hc, n2 = Pc * Dc, n3 = Dc * Rc, n4 = Hc * Dc;
        f32_to_bf16_kernel<<<(n1 + 255) / 256, 256, 0, stream>>>(in_w,  win_bf,  n1);
        f32_to_bf16_kernel<<<(n2 + 255) / 256, 256, 0, stream>>>(xp_w,  wxp_bf,  n2);
        f32_to_bf16_kernel<<<(n3 + 255) / 256, 256, 0, stream>>>(dt_w,  wdt_bf,  n3);
        f32_to_bf16_kernel<<<(n4 + 255) / 256, 256, 0, stream>>>(out_w, wout_bf, n4);
    }

    // 2) RMSNorm -> bf16
    rmsnorm_kernel<<<Mc, 256, 0, stream>>>(inputs, norm_w, xn_bf);

    // 3) in_proj: [M,H] x [2D,H]^T -> xg [M,2D] fp32
    gemm_wmma_bf16_kernel<<<dim3(Mc / BM, (2 * Dc) / BN), 256, 0, stream>>>(
        xn_bf, Hc, win_bf, Hc, in_b, nullptr, xgbuf, nullptr, 2 * Dc, Mc, 2 * Dc, Hc);

    // 4) depthwise conv + SiLU
    conv_silu_kernel<<<(Mc * Dc + 255) / 256, 256, 0, stream>>>(
        xgbuf, conv_w, conv_b, xc_f, xc_bf);

    // 5) x_proj: [M,D] x [96,D]^T -> proj [M,96] fp32 + bf16 (dt slice feeds GEMM)
    gemm_wmma_bf16_kernel<<<dim3(Mc / BM, (Pc + BN - 1) / BN), 256, 0, stream>>>(
        xc_bf, Dc, wxp_bf, Dc, xp_b, nullptr, proj_f, proj_bf, Pc, Mc, Pc, Dc);

    // 6) dt_proj: [M,64](lda=96) x [D,64]^T -> dlin [M,D] fp32 (softplus in scan)
    gemm_wmma_bf16_kernel<<<dim3(Mc / BM, Dc / BN), 256, 0, stream>>>(
        proj_bf, Pc, wdt_bf, Rc, dt_b, nullptr, dlin, nullptr, Dc, Mc, Dc, Rc);

    // 7) selective scan + gate -> y bf16
    scan_kernel<<<(Bc * Dc + 255) / 256, 256, 0, stream>>>(
        dlin, xc_f, xgbuf, proj_f, A_log, D_param, state0, y_bf);

    // 8) out_proj + residual: [M,D] x [H,D]^T + out_b + inputs -> d_out
    gemm_wmma_bf16_kernel<<<dim3(Mc / BM, Hc / BN), 256, 0, stream>>>(
        y_bf, Dc, wout_bf, Dc, out_b, inputs, out, nullptr, Hc, Mc, Hc, Dc);
}
